// GraphConv_47751446397508
// MI455X (gfx1250) — matrix operand order repaired and checked by reading, not compiled, not run on hardware
//
#include <hip/hip_runtime.h>
#include <stdint.h>

#define GN     10000      // nodes
#define GB     32         // batch
#define GF     1024       // features per node = D*B
#define GE     160000     // edges per support
#define GK     160        // GEMM K (NUM_METRICS)
#define GO     64         // out dim
#define GTILES ((GB * GN) / 16)   // 20000 row tiles

typedef __attribute__((ext_vector_type(16))) __bf16 v16bf;
typedef __attribute__((ext_vector_type(8)))  __bf16 v8bf;
typedef __attribute__((ext_vector_type(8)))  float  v8f;

// ---------------------------------------------------------------------------
// X0[node][b][d] = (bf16) inputs[b][node][d]
// [node,b,d] layout -> GEMM A rows are contiguous 8-elem bf16 runs (b128).
// bf16 X buffers: 5 x 20MB = 100MB, fully resident in the 192MB L2.
// ---------------------------------------------------------------------------
__global__ void build_x0_kernel(const float* __restrict__ in, __bf16* __restrict__ x0) {
    int i = blockIdx.x * blockDim.x + threadIdx.x;   // one 8-elem group
    if (i >= GN * (GF / 8)) return;
    int node = i >> 7;          // 128 groups per node
    int rb   = i & 127;
    int b    = rb >> 2;         // 4 groups per (node,b)
    int q    = rb & 3;
    const float* src = in + ((size_t)b * GN + node) * 32 + q * 8;
    float4 f0 = *(const float4*)(src);
    float4 f1 = *(const float4*)(src + 4);
    v8bf o;
    o[0] = (__bf16)f0.x; o[1] = (__bf16)f0.y; o[2] = (__bf16)f0.z; o[3] = (__bf16)f0.w;
    o[4] = (__bf16)f1.x; o[5] = (__bf16)f1.y; o[6] = (__bf16)f1.z; o[7] = (__bf16)f1.w;
    *(v8bf*)(x0 + (size_t)node * GF + rb * 8) = o;
}

// ---------------------------------------------------------------------------
// Seed accumulator: y = 0  or  y = -src (exact sign-bit flip on packed bf16),
// so the SPMM folds "2*spmm - x0_cur" into a single atomic pass.
// ---------------------------------------------------------------------------
__global__ void seed_kernel(uint4* __restrict__ y, const uint4* __restrict__ src,
                            int negate, int n8) {
    int i = blockIdx.x * blockDim.x + threadIdx.x;
    if (i >= n8) return;
    uint4 v;
    if (negate) {
        uint4 s = src[i];
        v.x = s.x ^ 0x80008000u; v.y = s.y ^ 0x80008000u;
        v.z = s.z ^ 0x80008000u; v.w = s.w ^ 0x80008000u;
    } else {
        v.x = 0u; v.y = 0u; v.z = 0u; v.w = 0u;
    }
    y[i] = v;
}

// ---------------------------------------------------------------------------
// COO SPMM scatter-add in bf16: y[rows[e]] += alpha*vals[e] * x[cols[e]]
// One block per edge (edge meta is blockIdx-uniform -> scalar loads).
// Hardware packed atomic: global_atomic_pk_add_bf16 (2 elems per atomic,
// device scope) -> half the atomic count and half the gather bytes of f32.
// ---------------------------------------------------------------------------
__global__ __launch_bounds__(128)
void spmm_atomic_kernel(const int* __restrict__ rows, const int* __restrict__ cols,
                        const float* __restrict__ vals, const __bf16* __restrict__ x,
                        __bf16* __restrict__ y, float alpha) {
    int e = blockIdx.x;
    int r = rows[e];
    int c = cols[e];
    float v = vals[e] * alpha;
    int t = threadIdx.x;                       // 128 threads * 8 bf16 = 1024
    v8bf xv = *(const v8bf*)(x + (size_t)c * GF + t * 8);
    const __bf16* yr = y + (size_t)r * GF;     // uniform -> SGPR base
    #pragma unroll
    for (int j = 0; j < 4; ++j) {
        float f0 = (float)xv[2 * j]     * v;
        float f1 = (float)xv[2 * j + 1] * v;
        __bf16 b0 = (__bf16)f0, b1 = (__bf16)f1;
        unsigned pk = ((unsigned)__builtin_bit_cast(unsigned short, b1) << 16)
                    |  (unsigned)__builtin_bit_cast(unsigned short, b0);
        unsigned voff = (unsigned)(t * 16 + j * 4);   // byte offset in row
        asm volatile("global_atomic_pk_add_bf16 %0, %1, %2 scope:SCOPE_DEV"
                     :: "v"(voff), "v"(pk), "s"(yr) : "memory");
    }
}

// ---------------------------------------------------------------------------
// GEMM: out[(b*GN+node), o] = sum_k A[.,k]*W[k,o] + bias[o]
//   A[(b,node), k=d*5+m] = Xm[node*GF + b*32 + d]  (bf16)
// K permuted to k' = m*32+d at W-staging time, so k-step s reads only Xs
// with contiguous d. W async-DMA'd into LDS (f32), converted once to bf16.
// Core op: v_wmma_f32_16x16x32_bf16, f32 accumulate, f32 output.
// All divergent global addressing uses 32-bit offsets off uniform SGPR bases
// (GVS mode) so the hot loop carries no 64-bit address VALU.
// ---------------------------------------------------------------------------
__global__ __launch_bounds__(256)
void gemm_wmma_kernel(const __bf16* __restrict__ X0, const __bf16* __restrict__ X1,
                      const __bf16* __restrict__ X2, const __bf16* __restrict__ X3,
                      const __bf16* __restrict__ X4, const float* __restrict__ W,
                      const float* __restrict__ bias, float* __restrict__ out,
                      int tiles) {
    __shared__ float  wstage[GK * GO];   // 40 KB raw f32 W (async landing zone)
    __shared__ __bf16 wh[GK * GO];       // 20 KB bf16, K-permuted
    __shared__ float  bsh[GO];

    // --- async-DMA W into LDS: 2560 x b128, 10 per thread ---
    {
        unsigned t = threadIdx.x;
        #pragma unroll
        for (int i = 0; i < 10; ++i) {
            unsigned idx    = t + i * 256u;                        // 16B chunk id
            unsigned ldsoff = (unsigned)(uintptr_t)(&wstage[idx * 4u]);
            unsigned goff   = idx * 16u;
            asm volatile("global_load_async_to_lds_b128 %0, %1, %2"
                         :: "v"(ldsoff), "v"(goff), "s"(W) : "memory");
        }
        asm volatile("s_wait_asynccnt 0" ::: "memory");
    }
    __syncthreads();

    // --- convert + K-permute:  wh[m*32+d][o] = (bf16) W[(d*5+m)*64 + o] ---
    for (int i = threadIdx.x; i < GK * GO; i += 256) {
        int kp = i >> 6;           // 0..159  (= m*32+d)
        int o  = i & 63;
        int m  = kp >> 5;
        int d  = kp & 31;
        wh[i] = (__bf16)wstage[(d * 5 + m) * GO + o];
    }
    if (threadIdx.x < GO) bsh[threadIdx.x] = bias[threadIdx.x];
    __syncthreads();

    const __bf16* bufs[5] = {X0, X1, X2, X3, X4};

    int wave   = blockIdx.x * (blockDim.x >> 5) + (threadIdx.x >> 5);
    int nwaves = gridDim.x * (blockDim.x >> 5);
    int lane   = threadIdx.x & 31;
    unsigned mrow  = (unsigned)(lane & 15);    // A row / B col / C col
    unsigned khalf = (unsigned)(lane >> 4);    // K half select

    // loop-invariant lane offsets (32-bit, hoisted)
    const unsigned aoff0 = 8u * khalf;                 // A lo-group elem offset
    const unsigned coff  = 8u * khalf * GO + mrow;     // C store elem offset
    const unsigned pfs   = (unsigned)nwaves * 16u * GF; // prefetch stride (elems)

    for (int rt = wave; rt < tiles; rt += nwaves) {
        int gr0   = rt * 16;                 // 16 | GN so tile stays in one b
        int b     = gr0 / GN;
        int node0 = gr0 % GN;
        // 32-bit divergent row offset off each uniform buffer base (GVS form)
        unsigned rowoff = (unsigned)(node0 + (int)mrow) * GF + (unsigned)b * 32u;

        v8f acc[4];
        #pragma unroll
        for (int nt = 0; nt < 4; ++nt) acc[nt] = {};

        #pragma unroll
        for (int s = 0; s < 5; ++s) {
            const __bf16* Xs = bufs[s];
            // A fragment (16x32 bf16): VGPR0-3 <- d = 8*khalf+0..7,
            //                          VGPR4-7 <- d = 16+8*khalf+0..7
            v8bf lo = *(const v8bf*)(Xs + rowoff + aoff0);        // one b128
            v8bf hi = *(const v8bf*)(Xs + rowoff + aoff0 + 16u);  // one b128
            v16bf a;
            #pragma unroll
            for (int j = 0; j < 8; ++j) { a[j] = lo[j]; a[8 + j] = hi[j]; }

            __builtin_prefetch(Xs + rowoff + pfs, 0, 0);

            #pragma unroll
            for (int nt = 0; nt < 4; ++nt) {
                // B fragment (32x16 bf16): lane group khalf holds K=8*khalf+j (j<8), 16+8*khalf+(j-8) (j>=8)
                const __bf16* wp = &wh[(s * 32 + 8 * (int)khalf) * GO + nt * 16 + (int)mrow];
                v16bf bf;
                #pragma unroll
                for (int j = 0; j < 16; ++j) bf[j] = wp[(j + (j >> 3) * 8) * GO];
                acc[nt] = __builtin_amdgcn_wmma_f32_16x16x32_bf16(
                    false, a, false, bf, (short)0, acc[nt], false, false);
            }
        }

        // C layout: lanes 0-15 -> M=v, lanes 16-31 -> M=v+8; N = mrow.
        // Uniform base per tile + one lane offset; v*GO + nt*16 folds into
        // the instruction immediate offset -> no per-store address VALU.
        float* ob = out + (size_t)gr0 * GO;          // wave-uniform (SGPR base)
        #pragma unroll
        for (int nt = 0; nt < 4; ++nt) {
            float bv = bsh[nt * 16 + mrow];
            #pragma unroll
            for (int v = 0; v < 8; ++v) {
                ob[coff + (unsigned)(v * GO + nt * 16)] = acc[nt][v] + bv;
            }
        }
    }
}

// ---------------------------------------------------------------------------
extern "C" void kernel_launch(void* const* d_in, const int* in_sizes, int n_in,
                              void* d_out, int out_size, void* d_ws, size_t ws_size,
                              hipStream_t stream) {
    const float* inputs   = (const float*)d_in[0];
    const int*   sup_rows = (const int*)d_in[1];
    const int*   sup_cols = (const int*)d_in[2];
    const float* sup_vals = (const float*)d_in[3];
    const float* weights  = (const float*)d_in[4];
    const float* bias     = (const float*)d_in[5];
    float*       out      = (float*)d_out;
    __bf16*      ws       = (__bf16*)d_ws;

    const size_t NF = (size_t)GN * GF;          // 10.24M elems per buffer (bf16)
    __bf16* X0 = ws + 0 * NF;
    __bf16* X1 = ws + 1 * NF;
    __bf16* X2 = ws + 2 * NF;
    __bf16* X3 = ws + 3 * NF;
    __bf16* X4 = ws + 4 * NF;

    const int n8 = (int)(NF / 8);               // uint4 (8 x bf16) chunks
    const int seed_blocks = (n8 + 255) / 256;

    // x0 in [node][b][d] bf16 layout
    build_x0_kernel<<<(GN * (GF / 8) + 255) / 256, 256, 0, stream>>>(inputs, X0);

    const int*   r0 = sup_rows;          const int*   r1 = sup_rows + GE;
    const int*   c0 = sup_cols;          const int*   c1 = sup_cols + GE;
    const float* v0 = sup_vals;          const float* v1 = sup_vals + GE;

    // T1 = S0 @ X0
    seed_kernel<<<seed_blocks, 256, 0, stream>>>((uint4*)X1, nullptr, 0, n8);
    spmm_atomic_kernel<<<GE, 128, 0, stream>>>(r0, c0, v0, X0, X1, 1.0f);
    // T2 = 2 * (S0 @ T1) - X0
    seed_kernel<<<seed_blocks, 256, 0, stream>>>((uint4*)X2, (const uint4*)X0, 1, n8);
    spmm_atomic_kernel<<<GE, 128, 0, stream>>>(r0, c0, v0, X1, X2, 2.0f);
    // T3 = S1 @ T1          (x0_cur is T1 after support 0)
    seed_kernel<<<seed_blocks, 256, 0, stream>>>((uint4*)X3, nullptr, 0, n8);
    spmm_atomic_kernel<<<GE, 128, 0, stream>>>(r1, c1, v1, X1, X3, 1.0f);
    // T4 = 2 * (S1 @ T3) - T1
    seed_kernel<<<seed_blocks, 256, 0, stream>>>((uint4*)X4, (const uint4*)X1, 1, n8);
    spmm_atomic_kernel<<<GE, 128, 0, stream>>>(r1, c1, v1, X3, X4, 2.0f);

    // out = A @ W + bias  via bf16 WMMA, f32 accumulate
    gemm_wmma_kernel<<<640, 256, 0, stream>>>(X0, X1, X2, X3, X4,
                                              weights, bias, out, GTILES);
}
